// KDSR_16655883174453
// MI455X (gfx1250) — matrix operand order repaired and checked by reading, not compiled, not run on hardware
//
#include <hip/hip_runtime.h>

typedef __attribute__((ext_vector_type(16))) _Float16 v16h;
typedef __attribute__((ext_vector_type(8)))  float    v8f;

#define N_BATCH 8
#define C 64
#define KTOT 576          // 9 taps * 64 ci
#define CHUNK 128
#define STRIPW 130
#define CPAD 72           // padded channel stride in LDS (36 dwords -> conflict-free)

// ---- CDNA5 async global->LDS copy (ASYNCcnt path), with safe fallback ------
#if defined(__has_builtin)
#  if __has_builtin(__builtin_amdgcn_global_load_async_to_lds_b128)
#    define HAS_ASYNC_LDS 1
#  endif
#  if __has_builtin(__builtin_amdgcn_s_wait_asynccnt)
#    define HAS_WAIT_ASYNC 1
#  endif
#endif
#ifndef HAS_ASYNC_LDS
#  define HAS_ASYNC_LDS 0
#endif
#ifndef HAS_WAIT_ASYNC
#  define HAS_WAIT_ASYNC 0
#endif

// Builtin param type per hipcc diagnostic: vector_size(16) int, AS1 (global).
typedef int v4i_vs __attribute__((vector_size(16)));
typedef __attribute__((address_space(1))) v4i_vs g_v4i;   // global b128 chunk
typedef __attribute__((address_space(3))) v4i_vs l_v4i;   // LDS b128 chunk

union ABFrag { v16h v; uint4 u[2]; };

// ---------------------------------------------------------------------------
// Implicit-GEMM 3x3 conv, Cin=64 (f16 NHWC input), Cout multiple of 64.
// f16 WMMA, f32 accumulate. Optional residual add (f32 NCHW), optional f32
// NCHW main output, optional f16 NHWC "shadow" output (feeds the next conv),
// optional fused 2x pixel-shuffle on the output coordinates.
// ---------------------------------------------------------------------------
__global__ __launch_bounds__(256) void conv64_wmma(
    const _Float16* __restrict__ in,   // [N][H][W][64] f16 NHWC
    const _Float16* __restrict__ wh,   // [coutTotal][9*64], k = tap*64 + ci
    const float* __restrict__ bias,    // [coutTotal]
    const float* __restrict__ res,     // [N][coutTotal][H][W] f32 or null
    float* __restrict__ out,           // f32 main output (NCHW / shuffled NCHW) or null
    _Float16* __restrict__ shadow,     // f16 NHWC copy in output coords, or null
    int H, int W, int coutTotal, int coGroups, int shuffle)
{
    __shared__ __align__(16) _Float16 strip[3 * STRIPW * CPAD];

    const int b  = blockIdx.z / coGroups;
    const int cg = blockIdx.z % coGroups;
    const int y  = blockIdx.y;
    const int x0 = blockIdx.x * CHUNK;
    const int tid = threadIdx.x;

    // ---- Phase 1a: zero the strip (covers SAME-padding borders).
    {
        uint4 z; z.x = z.y = z.z = z.w = 0u;
        uint4* sp = (uint4*)strip;
        for (int i = tid; i < (3 * STRIPW * CPAD) / 8; i += 256) sp[i] = z;
    }
    __syncthreads();

    // ---- Phase 1b: stage interior as 16B chunks (8 ch) per lane via the
    // CDNA5 async global->LDS path (ASYNCcnt). NHWC f16 makes each (row,col)
    // position 128 contiguous bytes.
    for (int idx = tid; idx < 3 * STRIPW * 8; idx += 256) {
        int ck  = idx & 7;
        int pos = idx >> 3;
        int r   = pos / STRIPW;
        int col = pos - r * STRIPW;
        int gy = y + r - 1;
        int gx = x0 + col - 1;
        if ((unsigned)gy < (unsigned)H && (unsigned)gx < (unsigned)W) {
            const _Float16* g = in + (((size_t)b * H + gy) * W + gx) * 64 + ck * 8;
            _Float16* l = &strip[(r * STRIPW + col) * CPAD + ck * 8];
#if HAS_ASYNC_LDS
            __builtin_amdgcn_global_load_async_to_lds_b128(
                (g_v4i*)g, (l_v4i*)l, 0, 0);
#else
            *(uint4*)l = *(const uint4*)g;
#endif
        }
    }
#if HAS_ASYNC_LDS
#  if HAS_WAIT_ASYNC
    __builtin_amdgcn_s_wait_asynccnt(0);
#  else
    asm volatile("s_wait_asynccnt 0" ::: "memory");
#  endif
#endif
    __syncthreads();

    // ---- Phase 2: WMMA GEMM. Wave -> one 16-cout tile x four 16-pixel tiles.
    const int wv   = tid >> 5;
    const int lane = tid & 31;
    const int row  = lane & 15;      // M (A), N (B/C)
    const int hs   = lane >> 4;      // half-select
    const int coTile = wv >> 1;      // 0..3
    const int pHalf  = wv & 1;       // 0..1
    const int coBase = cg * 64 + coTile * 16;

    v8f acc[4] = {};

    const _Float16* wrow = wh + (size_t)(coBase + row) * KTOT;

    #pragma unroll
    for (int t = 0; t < 9; ++t) {
        const int dy = t / 3, dx = t % 3;
        #pragma unroll
        for (int c2 = 0; c2 < 2; ++c2) {
            // A fragment: K = hs*8 + i (i<8), 16 + hs*8 + (i-8) (i>=8); M = row
            ABFrag a;
            const _Float16* wp = wrow + t * 64 + c2 * 32 + hs * 8;
            a.u[0] = *(const uint4*)(wp);
            a.u[1] = *(const uint4*)(wp + 16);

            // Issue all four B-fragment loads (8x ds_load_b128) into distinct
            // registers BEFORE any WMMA, so the backend can use partial
            // s_wait_dscnt counts and overlap DS loads with matrix ops.
            ABFrag bf[4];
            #pragma unroll
            for (int p = 0; p < 4; ++p) {
                const int px = (pHalf * 4 + p) * 16 + row;   // pixel = N
                // B fragment: K = hs*8 + i (i<8), 16 + hs*8 + (i-8) (i>=8); N = row
                const _Float16* bp =
                    &strip[(dy * STRIPW + px + dx) * CPAD + c2 * 32 + hs * 8];
                bf[p].u[0] = *(const uint4*)(bp);
                bf[p].u[1] = *(const uint4*)(bp + 16);
            }
            #pragma unroll
            for (int p = 0; p < 4; ++p) {
                acc[p] = __builtin_amdgcn_wmma_f32_16x16x32_f16(
                    false, a.v, false, bf[p].v, (short)0, acc[p], false, false);
            }
        }
    }

    // ---- Epilogue: bias (+residual), f32 main store and/or f16 NHWC shadow,
    //      optionally through fused 2x pixel-shuffle coordinates.
    const int outC = coutTotal >> 2;
    const int OH = H * 2, OW = W * 2;
    #pragma unroll
    for (int p = 0; p < 4; ++p) {
        const int px = (pHalf * 4 + p) * 16 + row;
        const int x = x0 + px;
        #pragma unroll
        for (int r = 0; r < 8; ++r) {
            const int m  = hs * 8 + r;       // C/D: M = r + 8*hs, N = row
            const int co = coBase + m;
            float v = acc[p][r] + bias[co];
            if (shuffle) {
                const int o  = co >> 2;
                const int oy = 2 * y + ((co >> 1) & 1);
                const int ox = 2 * x + (co & 1);
                if (out)
                    out[(((size_t)b * outC + o) * OH + oy) * OW + ox] = v;
                if (shadow)
                    shadow[(((size_t)b * OH + oy) * OW + ox) * 64 + o] = (_Float16)v;
            } else {
                if (res) v += res[(((size_t)b * coutTotal + co) * H + y) * W + x];
                if (out)
                    out[(((size_t)b * coutTotal + co) * H + y) * W + x] = v;
                if (shadow)
                    shadow[(((size_t)b * H + y) * W + x) * 64 + co] = (_Float16)v;
            }
        }
    }
}

// ---------------------------------------------------------------------------
// Weight repack f32 [cout][ci][3][3] -> f16 [cout][tap*64+ci]
// ---------------------------------------------------------------------------
__global__ void wcvt_kernel(const float* __restrict__ src, _Float16* __restrict__ dst,
                            int total)
{
    int idx = blockIdx.x * 256 + threadIdx.x;
    if (idx >= total) return;
    int co = idx / KTOT;
    int k  = idx - co * KTOT;
    int t  = k >> 6;
    int ci = k & 63;
    dst[idx] = (_Float16)src[(size_t)co * KTOT + ci * 9 + t];
}

// ---------------------------------------------------------------------------
// Head: (x - mean) -> conv 3->64 + bias. f32 NCHW output (residual anchor).
// ---------------------------------------------------------------------------
__global__ void head_kernel(const float* __restrict__ x, const float* __restrict__ w,
                            const float* __restrict__ bias, float* __restrict__ out)
{
    const float mean[3] = {0.4488f, 0.4371f, 0.404f};
    int idx = blockIdx.x * 256 + threadIdx.x;
    if (idx >= N_BATCH * C * 128 * 128) return;
    int xx = idx & 127;
    int yy = (idx >> 7) & 127;
    int co = (idx >> 14) & 63;
    int b  = idx >> 20;
    float s = bias[co];
    #pragma unroll
    for (int c = 0; c < 3; ++c)
        #pragma unroll
        for (int dy = 0; dy < 3; ++dy) {
            int gy = yy + dy - 1;
            if ((unsigned)gy >= 128u) continue;
            #pragma unroll
            for (int dx = 0; dx < 3; ++dx) {
                int gx = xx + dx - 1;
                if ((unsigned)gx >= 128u) continue;
                float v = x[(((size_t)b * 3 + c) * 128 + gy) * 128 + gx] - mean[c];
                s += w[((co * 3 + c) * 3 + dy) * 3 + dx] * v;
            }
        }
    out[idx] = s;
}

// ---------------------------------------------------------------------------
// Dynamic kernel MLP: k = leaky(kv @ w1^T) @ w2^T, per block i (blockIdx.x).
// ---------------------------------------------------------------------------
__global__ void dynk_kernel(const float* __restrict__ kv, const float* __restrict__ w1,
                            const float* __restrict__ w2, float* __restrict__ dk)
{
    const int i = blockIdx.x;
    const int tid = threadIdx.x;
    __shared__ float hid[N_BATCH * C];
    for (int j = tid; j < N_BATCH * C; j += 256) {
        int b = j >> 6, h = j & 63;
        const float* wr = w1 + ((size_t)i * C + h) * C;
        const float* kr = kv + b * C;
        float s = 0.f;
        for (int c = 0; c < C; ++c) s += kr[c] * wr[c];
        hid[j] = s > 0.f ? s : 0.1f * s;
    }
    __syncthreads();
    for (int m = tid; m < N_BATCH * KTOT; m += 256) {
        int b = m / KTOT, mm = m - b * KTOT;
        const float* wr = w2 + ((size_t)i * KTOT + mm) * C;
        const float* hr = hid + b * C;
        float s = 0.f;
        for (int j = 0; j < C; ++j) s += hr[j] * wr[j];
        dk[(size_t)i * N_BATCH * KTOT + m] = s;
    }
}

// ---------------------------------------------------------------------------
// Dynamic depthwise 3x3 conv + fused LeakyReLU(0.1).
// Input f32 NCHW, output f16 NHWC (direct conv-feed format, half the traffic).
// ---------------------------------------------------------------------------
__global__ void ddc_leaky_kernel(const float* __restrict__ in, const float* __restrict__ dk,
                                 _Float16* __restrict__ out)
{
    int idx = blockIdx.x * 256 + threadIdx.x;
    if (idx >= N_BATCH * C * 128 * 128) return;
    int c  = idx & 63;
    int xx = (idx >> 6) & 127;
    int yy = (idx >> 13) & 127;
    int b  = idx >> 20;
    const float* kk   = dk + ((size_t)b * C + c) * 9;
    const float* base = in + (((size_t)b * C + c) * 128) * 128;
    float s = 0.f;
    #pragma unroll
    for (int dy = 0; dy < 3; ++dy) {
        int gy = yy + dy - 1;
        if ((unsigned)gy >= 128u) continue;
        #pragma unroll
        for (int dx = 0; dx < 3; ++dx) {
            int gx = xx + dx - 1;
            if ((unsigned)gx >= 128u) continue;
            s += kk[dy * 3 + dx] * base[gy * 128 + gx];
        }
    }
    s = s > 0.f ? s : 0.1f * s;
    out[(((size_t)b * 128 + yy) * 128 + xx) * 64 + c] = (_Float16)s;
}

// ---------------------------------------------------------------------------
// Tail: conv 64->3 @512x512 + bias + re-add mean. One thread per pixel.
// ---------------------------------------------------------------------------
__global__ void tail_kernel(const float* __restrict__ in, const float* __restrict__ w,
                            const float* __restrict__ bias, float* __restrict__ out)
{
    const int HW = 512 * 512;
    int idx = blockIdx.x * 256 + threadIdx.x;
    if (idx >= N_BATCH * HW) return;
    int xx = idx & 511;
    int yy = (idx >> 9) & 511;
    int b  = idx >> 18;
    const float mean[3] = {0.4488f, 0.4371f, 0.404f};
    float a0 = 0.f, a1 = 0.f, a2 = 0.f;
    for (int ci = 0; ci < C; ++ci) {
        const float* base = in + (((size_t)b * C + ci) * 512) * 512;
        #pragma unroll
        for (int dy = 0; dy < 3; ++dy) {
            int gy = yy + dy - 1;
            if ((unsigned)gy >= 512u) continue;
            #pragma unroll
            for (int dx = 0; dx < 3; ++dx) {
                int gx = xx + dx - 1;
                if ((unsigned)gx >= 512u) continue;
                float v = base[gy * 512 + gx];
                a0 += w[((0 * C + ci) * 3 + dy) * 3 + dx] * v;
                a1 += w[((1 * C + ci) * 3 + dy) * 3 + dx] * v;
                a2 += w[((2 * C + ci) * 3 + dy) * 3 + dx] * v;
            }
        }
    }
    size_t o = (size_t)b * 3 * HW + (size_t)yy * 512 + xx;
    out[o]          = a0 + bias[0] + mean[0];
    out[o + HW]     = a1 + bias[1] + mean[1];
    out[o + 2 * HW] = a2 + bias[2] + mean[2];
}

// ---------------------------------------------------------------------------
extern "C" void kernel_launch(void* const* d_in, const int* in_sizes, int n_in,
                              void* d_out, int out_size, void* d_ws, size_t ws_size,
                              hipStream_t stream)
{
    (void)in_sizes; (void)n_in; (void)out_size; (void)ws_size;
    const float* x       = (const float*)d_in[0];
    const float* k_v     = (const float*)d_in[1];
    const float* head_w  = (const float*)d_in[2];
    const float* head_b  = (const float*)d_in[3];
    const float* kmlp1   = (const float*)d_in[4];
    const float* kmlp2   = (const float*)d_in[5];
    const float* convw   = (const float*)d_in[6];
    const float* convb   = (const float*)d_in[7];
    const float* bodyc_w = (const float*)d_in[8];
    const float* bodyc_b = (const float*)d_in[9];
    const float* up1_w   = (const float*)d_in[10];
    const float* up1_b   = (const float*)d_in[11];
    const float* up2_w   = (const float*)d_in[12];
    const float* up2_b   = (const float*)d_in[13];
    const float* tail_w  = (const float*)d_in[14];
    const float* tail_b  = (const float*)d_in[15];
    float* out = (float*)d_out;

    char* ws = (char*)d_ws;
    size_t off = 0;
    auto alloc = [&](size_t bytes) {
        void* p = ws + off;
        off = (off + bytes + 255) & ~((size_t)255);
        return p;
    };
    _Float16* whB  = (_Float16*)alloc((size_t)512 * KTOT * 2);  // 8 blocks x 64
    _Float16* whC  = (_Float16*)alloc((size_t)64  * KTOT * 2);
    _Float16* whU1 = (_Float16*)alloc((size_t)256 * KTOT * 2);
    _Float16* whU2 = (_Float16*)alloc((size_t)256 * KTOT * 2);
    float* dk = (float*)alloc((size_t)8 * N_BATCH * KTOT * 4);

    const size_t hw128 = (size_t)N_BATCH * 128 * 128;
    float*    bufX  = (float*)   alloc(hw128 * C * 4);   // head out (residual anchor)
    float*    bufA  = (float*)   alloc(hw128 * C * 4);   // residual ping
    float*    bufB  = (float*)   alloc(hw128 * C * 4);   // residual pong
    _Float16* bufT  = (_Float16*)alloc(hw128 * C * 2);   // ddc out, f16 NHWC
    _Float16* shA   = (_Float16*)alloc(hw128 * C * 2);   // block7 shadow -> bodyc in
    _Float16* shB   = (_Float16*)alloc(hw128 * C * 2);   // bodyc shadow -> up1 in
    _Float16* sh256 = (_Float16*)alloc(hw128 * 4 * C * 2);  // up1 shadow (256^2 NHWC)
    float*    buf512 = (float*)  alloc(hw128 * 16 * C * 4); // up2 out (512^2 NCHW f32)

    // Weight repack to f16 (tap-major K order for contiguous A-fragment loads)
    wcvt_kernel<<<(512 * KTOT + 255) / 256, 256, 0, stream>>>(convw,   whB,  512 * KTOT);
    wcvt_kernel<<<(64  * KTOT + 255) / 256, 256, 0, stream>>>(bodyc_w, whC,  64  * KTOT);
    wcvt_kernel<<<(256 * KTOT + 255) / 256, 256, 0, stream>>>(up1_w,   whU1, 256 * KTOT);
    wcvt_kernel<<<(256 * KTOT + 255) / 256, 256, 0, stream>>>(up2_w,   whU2, 256 * KTOT);

    // Dynamic per-channel kernels (tiny MLP)
    dynk_kernel<<<8, 256, 0, stream>>>(k_v, kmlp1, kmlp2, dk);

    // Head conv (3->64) with mean subtraction
    head_kernel<<<(N_BATCH * C * 128 * 128) / 256, 256, 0, stream>>>(x, head_w, head_b, bufX);

    // 8 residual blocks: ddc+leaky (f16 NHWC) -> conv64 (+bias +residual)
    float* cur = bufX;
    for (int i = 0; i < 8; ++i) {
        ddc_leaky_kernel<<<(N_BATCH * C * 128 * 128) / 256, 256, 0, stream>>>(
            cur, dk + (size_t)i * N_BATCH * KTOT, bufT);
        float* nxt = (cur == bufA) ? bufB : bufA;
        conv64_wmma<<<dim3(1, 128, N_BATCH), 256, 0, stream>>>(
            bufT, whB + (size_t)i * 64 * KTOT, convb + i * 64, cur,
            (i == 7) ? nullptr : nxt,          // last block: f32 out unused
            (i == 7) ? shA : nullptr,          // last block: emit f16 shadow
            128, 128, 64, 1, 0);
        if (i < 7) cur = nxt;
    }

    // Body conv + long skip (residual = head out); emit only the f16 shadow
    conv64_wmma<<<dim3(1, 128, N_BATCH), 256, 0, stream>>>(
        shA, whC, bodyc_b, bufX, nullptr, shB, 128, 128, 64, 1, 0);

    // up1: 64->256 @128x128, fused pixel-shuffle, f16 NHWC shadow only
    conv64_wmma<<<dim3(1, 128, N_BATCH * 4), 256, 0, stream>>>(
        shB, whU1, up1_b, nullptr, nullptr, sh256, 128, 128, 256, 4, 1);

    // up2: 64->256 @256x256, fused pixel-shuffle -> f32 [8][64][512][512]
    conv64_wmma<<<dim3(2, 256, N_BATCH * 4), 256, 0, stream>>>(
        sh256, whU2, up2_b, nullptr, buf512, nullptr, 256, 256, 256, 4, 1);

    // tail: 64->3 @512x512, + mean re-add
    tail_kernel<<<(N_BATCH * 512 * 512 + 255) / 256, 256, 0, stream>>>(
        buf512, tail_w, tail_b, out);
}
